// GptOssDecoderLayer_55542517071893
// MI455X (gfx1250) — matrix operand order repaired and checked by reading, not compiled, not run on hardware
//
#include <hip/hip_runtime.h>
#include <hip/hip_bf16.h>

// ---------------------------------------------------------------------------
// GPT-OSS decoder layer for MI455X (gfx1250, wave32, WMMA).
// Heavy GEMMs: double-buffered LDS tiles, float4 global loads, f32->f16
// convert on staging, 128-bit LDS fragment loads, v_wmma_f32_16x16x32_f16.
// ---------------------------------------------------------------------------

typedef __attribute__((ext_vector_type(16))) _Float16 v16h;
typedef __attribute__((ext_vector_type(8)))  _Float16 v8h;
typedef __attribute__((ext_vector_type(4)))  _Float16 v4h;
typedef __attribute__((ext_vector_type(8)))  float    v8f;
typedef __attribute__((ext_vector_type(4)))  float    v4f;

#define ALPHA 1.702f
#define LIMIT 7.0f
#define EPS   1e-5f
#define THETA 150000.0f
#define S_LEN 1024
#define HID   2880
#define NH    64
#define NKV   8
#define HD    64
#define QDIM  (NH * HD)   // 4096
#define KVDIM (NKV * HD)  // 512
#define NEXP  8
#define IDIM  2880

#define LDG   40   // GEMM LDS row stride (halves); 80B, 16B-aligned
#define LDQ   72   // attention Q/K LDS row stride; 144B, 16B-aligned
#define LDP   40   // attention P/Vt LDS row stride

__device__ __forceinline__ v8f wmma16x16x32(v16h a, v16h b, v8f c) {
  return __builtin_amdgcn_wmma_f32_16x16x32_f16(
      /*neg_a=*/false, a, /*neg_b=*/false, b,
      /*c_mod=*/(short)0, c, /*reuse_a=*/false, /*reuse_b=*/false);
}

// A fragment 16x32 f16, tile stored row-major [m][k], row stride ld (halves).
// Per-lane data is two contiguous 16B chunks: halves [kb..kb+7], [kb+16..kb+23]
// with kb = 0 (lanes 0-15) or 8 (lanes 16-31).  -> 2 x ds_load_b128.
__device__ __forceinline__ v16h load_frag_a(const _Float16* s, int ld, int lane) {
  int m  = lane & 15;
  int kb = (lane < 16) ? 0 : 8;
  const _Float16* r = s + m * ld + kb;
  v8h lo = *(const v8h*)(r);
  v8h hi = *(const v8h*)(r + 16);
  return __builtin_shufflevector(lo, hi, 0, 1, 2, 3, 4, 5, 6, 7,
                                 8, 9, 10, 11, 12, 13, 14, 15);
}

// B fragment 32x16 f16 (KxN), tile stored [n][k], row stride ld (halves).
// Per-lane data is two contiguous 16B chunks: halves [kb..kb+7], [kb+16..kb+23], kb = 0 / 8. -> 2 x b128.
__device__ __forceinline__ v16h load_frag_b(const _Float16* s, int ld, int lane) {
  int n  = lane & 15;
  int kb = (lane < 16) ? 0 : 8;
  const _Float16* r = s + n * ld + kb;
  v8h lo = *(const v8h*)(r);
  v8h hi = *(const v8h*)(r + 16);
  return __builtin_shufflevector(lo, hi, 0, 1, 2, 3, 4, 5, 6, 7,
                                 8, 9, 10, 11, 12, 13, 14, 15);
}

__device__ __forceinline__ v4h cvt4(v4f t) {
  v4h h;
  h[0] = (_Float16)t[0]; h[1] = (_Float16)t[1];
  h[2] = (_Float16)t[2]; h[3] = (_Float16)t[3];
  return h;
}

// ---------------------------------------------------------------------------
// Generic WMMA GEMM: C[M,N] = A[M,K] * B[N,K]^T (+bias / +resid / scaled acc).
// BM=128, BN=64, BK=32; 256 threads = 8 waves (4Mx2N), 32x32 per wave.
// Double-buffered LDS; next tile's global_load_b128s issued before the WMMAs.
// flags: 1=add bias[n], 2=add resid[m*N+n], 4=C += row_scale[m*ss]*val
// ---------------------------------------------------------------------------
__global__ __launch_bounds__(256) void gemm_wmma_kernel(
    const float* __restrict__ A, const float* __restrict__ B,
    const float* __restrict__ bias, const float* __restrict__ resid,
    const float* __restrict__ row_scale, float* __restrict__ C,
    int M, int N, int K, int scale_stride, int flags) {
  __shared__ __align__(16) _Float16 sA[2][128][LDG];
  __shared__ __align__(16) _Float16 sB[2][64][LDG];

  const int tid  = threadIdx.x;
  const int lane = tid & 31;
  const int wave = tid >> 5;
  const int wm   = (wave >> 1) * 32;
  const int wn   = (wave & 1) * 32;
  const int mBase = blockIdx.y * 128;
  const int nBase = blockIdx.x * 64;

  const float* Abase = A + (size_t)mBase * K;
  const float* Bbase = B + (size_t)nBase * K;

  // per-thread staging coordinates (float4 granularity, 8 chunks per 32-col row)
  const int ra = tid >> 3, ca = (tid & 7) * 4;   // + j*32 rows for A (4 chunks)
  const int rb = tid >> 3, cb = (tid & 7) * 4;   // + j*32 rows for B (2 chunks)

  v8f acc[2][2] = {};

  // prologue: stage k0=0 into buffer 0
#pragma unroll
  for (int j = 0; j < 4; ++j) {
    v4f t = *(const v4f*)&Abase[(size_t)(ra + j * 32) * K + ca];
    *(v4h*)&sA[0][ra + j * 32][ca] = cvt4(t);
  }
#pragma unroll
  for (int j = 0; j < 2; ++j) {
    v4f t = *(const v4f*)&Bbase[(size_t)(rb + j * 32) * K + cb];
    *(v4h*)&sB[0][rb + j * 32][cb] = cvt4(t);
  }
  __syncthreads();

  for (int k0 = 0, buf = 0; k0 < K; k0 += 32, buf ^= 1) {
    const bool hasNext = (k0 + 32) < K;
    v4f pa[4], pb[2];
    if (hasNext) {
#pragma unroll
      for (int j = 0; j < 4; ++j)
        pa[j] = *(const v4f*)&Abase[(size_t)(ra + j * 32) * K + (k0 + 32) + ca];
#pragma unroll
      for (int j = 0; j < 2; ++j)
        pb[j] = *(const v4f*)&Bbase[(size_t)(rb + j * 32) * K + (k0 + 32) + cb];
    }

    v16h af[2], bf[2];
    af[0] = load_frag_a(&sA[buf][wm][0], LDG, lane);
    af[1] = load_frag_a(&sA[buf][wm + 16][0], LDG, lane);
    bf[0] = load_frag_b(&sB[buf][wn][0], LDG, lane);
    bf[1] = load_frag_b(&sB[buf][wn + 16][0], LDG, lane);
#pragma unroll
    for (int im = 0; im < 2; ++im)
#pragma unroll
      for (int in = 0; in < 2; ++in)
        acc[im][in] = wmma16x16x32(af[im], bf[in], acc[im][in]);

    if (hasNext) {
      int nb = buf ^ 1;
#pragma unroll
      for (int j = 0; j < 4; ++j) *(v4h*)&sA[nb][ra + j * 32][ca] = cvt4(pa[j]);
#pragma unroll
      for (int j = 0; j < 2; ++j) *(v4h*)&sB[nb][rb + j * 32][cb] = cvt4(pb[j]);
    }
    __syncthreads();
  }

  const int off = (lane < 16) ? 0 : 8;
  const int col = lane & 15;
#pragma unroll
  for (int im = 0; im < 2; ++im)
#pragma unroll
    for (int in = 0; in < 2; ++in)
#pragma unroll
      for (int vv = 0; vv < 8; ++vv) {
        int m = mBase + wm + im * 16 + vv + off;
        int n = nBase + wn + in * 16 + col;
        float val = acc[im][in][vv];
        if (flags & 1) val += bias[n];
        if (flags & 2) val += resid[(size_t)m * N + n];
        if (flags & 4)
          C[(size_t)m * N + n] += row_scale[(size_t)m * scale_stride] * val;
        else
          C[(size_t)m * N + n] = val;
      }
}

// ---------------------------------------------------------------------------
// RMSNorm: one block per row, float4 loads.
// ---------------------------------------------------------------------------
__global__ __launch_bounds__(256) void rmsnorm_kernel(
    const float* __restrict__ x, const float* __restrict__ w,
    float* __restrict__ y, int H) {
  int row = blockIdx.x;
  const float* xr = x + (size_t)row * H;
  const int H4 = H >> 2;
  float ss = 0.f;
  for (int i = threadIdx.x; i < H4; i += 256) {
    v4f v = *(const v4f*)&xr[i * 4];
    ss += v[0] * v[0] + v[1] * v[1] + v[2] * v[2] + v[3] * v[3];
  }
  __shared__ float red[256];
  red[threadIdx.x] = ss;
  __syncthreads();
  for (int s = 128; s > 0; s >>= 1) {
    if (threadIdx.x < s) red[threadIdx.x] += red[threadIdx.x + s];
    __syncthreads();
  }
  float scale = rsqrtf(red[0] / (float)H + EPS);
  for (int i = threadIdx.x; i < H4; i += 256) {
    v4f v = *(const v4f*)&xr[i * 4];
    v4f g = *(const v4f*)&w[i * 4];
    v4f o;
    o[0] = v[0] * scale * g[0]; o[1] = v[1] * scale * g[1];
    o[2] = v[2] * scale * g[2]; o[3] = v[3] * scale * g[3];
    *(v4f*)&y[(size_t)row * H + i * 4] = o;
  }
}

// ---------------------------------------------------------------------------
// RoPE (rotate-half) applied in-place to q [S,4096] and k [S,512].
// ---------------------------------------------------------------------------
__global__ void rope_kernel(float* __restrict__ q, float* __restrict__ k) {
  int idx = blockIdx.x * blockDim.x + threadIdx.x;
  const int total = S_LEN * (NH + NKV) * 32;
  if (idx >= total) return;
  int d    = idx & 31;
  int rest = idx >> 5;
  int hh   = rest % (NH + NKV);
  int t    = rest / (NH + NKV);
  float inv = __powf(THETA, -((float)(2 * d)) * (1.0f / 64.0f));
  float ang = (float)t * inv;
  float sn, cs;
  __sincosf(ang, &sn, &cs);
  float* buf; int ld, base;
  if (hh < NH) { buf = q; ld = QDIM;  base = hh * HD; }
  else         { buf = k; ld = KVDIM; base = (hh - NH) * HD; }
  size_t o = (size_t)t * ld + base;
  float x1 = buf[o + d], x2 = buf[o + d + 32];
  buf[o + d]      = x1 * cs - x2 * sn;
  buf[o + d + 32] = x1 * sn + x2 * cs;
}

// ---------------------------------------------------------------------------
// Flash attention with sinks + causal mask + GQA.
// Grid: (S/64, NH); 128 threads = 4 waves; wave w owns rows w*16..w*16+15.
// Online softmax seeded with the sink logit (m=sink, l=1); sink column is
// dropped from the numerator, matching the reference exactly.
// V is staged transposed (sVt[d][key]) so P*V B-fragments are 128-bit loads.
// ---------------------------------------------------------------------------
__global__ __launch_bounds__(128) void attn_kernel(
    const float* __restrict__ q, const float* __restrict__ kbuf,
    const float* __restrict__ vbuf, const float* __restrict__ sinks,
    float* __restrict__ out) {
  const int qb   = blockIdx.x;
  const int head = blockIdx.y;
  const int kvh  = head >> 3;
  const int tid  = threadIdx.x;
  const int lane = tid & 31;
  const int w    = tid >> 5;
  const int qBase = qb * 64;

  __shared__ __align__(16) _Float16 sQ[64][LDQ];
  __shared__ __align__(16) _Float16 sK[32][LDQ];
  __shared__ __align__(16) _Float16 sVt[64][LDP];   // [d][key]
  __shared__ __align__(16) _Float16 sP[4][16][LDP];

  // stage Q (64x64, float4 granularity: 1024 chunks / 128 threads = 8)
#pragma unroll
  for (int j = 0; j < 8; ++j) {
    int i = tid + j * 128;
    int r = i >> 4, col = (i & 15) * 4;
    v4f t = *(const v4f*)&q[(size_t)(qBase + r) * QDIM + head * HD + col];
    *(v4h*)&sQ[r][col] = cvt4(t);
  }
  __syncthreads();

  const int off = (lane < 16) ? 0 : 8;
  const int col = lane & 15;
  const float snk = sinks[head];
  float mi[8], li[8];
  v8f o[4] = {};
#pragma unroll
  for (int vv = 0; vv < 8; ++vv) { mi[vv] = snk; li[vv] = 1.0f; }

  const int kEnd = qBase + 64;
  for (int kb0 = 0; kb0 < kEnd; kb0 += 32) {
    // stage K (32x64) and V transposed: 512 float4 chunks each / 128 = 4
#pragma unroll
    for (int j = 0; j < 4; ++j) {
      int i = tid + j * 128;
      int r = i >> 4, c = (i & 15) * 4;   // r = key row, c = dim base
      v4f tk = *(const v4f*)&kbuf[(size_t)(kb0 + r) * KVDIM + kvh * HD + c];
      *(v4h*)&sK[r][c] = cvt4(tk);
      v4f tv = *(const v4f*)&vbuf[(size_t)(kb0 + r) * KVDIM + kvh * HD + c];
      sVt[c][r]     = (_Float16)tv[0];
      sVt[c + 1][r] = (_Float16)tv[1];
      sVt[c + 2][r] = (_Float16)tv[2];
      sVt[c + 3][r] = (_Float16)tv[3];
    }
    __syncthreads();

    // S = Q (16x64) x Kblk^T (64x32): two 16x16 accumulators, K split in 2
    v8f sc[2] = {};
#pragma unroll
    for (int ks = 0; ks < 64; ks += 32) {
      v16h aq = load_frag_a(&sQ[w * 16][ks], LDQ, lane);
#pragma unroll
      for (int nt = 0; nt < 2; ++nt) {
        v16h bk = load_frag_b(&sK[nt * 16][ks], LDQ, lane);
        sc[nt] = wmma16x16x32(aq, bk, sc[nt]);
      }
    }

    // online softmax (rows live across the 16 lanes of a half-wave)
#pragma unroll
    for (int vv = 0; vv < 8; ++vv) {
      int m = vv + off;
      int t = qBase + w * 16 + m;
      float s0 = sc[0][vv] * 0.125f;  // HEAD_DIM^-0.5
      float s1 = sc[1][vv] * 0.125f;
      bool msk0 = (kb0 + col) > t;
      bool msk1 = (kb0 + 16 + col) > t;
      if (msk0) s0 = -1e30f;
      if (msk1) s1 = -1e30f;
      float rm = fmaxf(s0, s1);
#pragma unroll
      for (int d = 1; d < 16; d <<= 1) rm = fmaxf(rm, __shfl_xor(rm, d, 16));
      float mnew  = fmaxf(mi[vv], rm);
      float scale = __expf(mi[vv] - mnew);
      float p0 = msk0 ? 0.f : __expf(s0 - mnew);
      float p1 = msk1 ? 0.f : __expf(s1 - mnew);
      float rs = p0 + p1;
#pragma unroll
      for (int d = 1; d < 16; d <<= 1) rs += __shfl_xor(rs, d, 16);
      li[vv] = li[vv] * scale + rs;
      mi[vv] = mnew;
      sP[w][m][col]      = (_Float16)p0;
      sP[w][m][16 + col] = (_Float16)p1;
#pragma unroll
      for (int j = 0; j < 4; ++j) o[j][vv] *= scale;
    }

    // O += P (16x32) x V (32x64); B-fragments from transposed V tile
    v16h ap = load_frag_a(&sP[w][0][0], LDP, lane);
#pragma unroll
    for (int j = 0; j < 4; ++j) {
      v16h bv = load_frag_b(&sVt[j * 16][0], LDP, lane);
      o[j] = wmma16x16x32(ap, bv, o[j]);
    }
    __syncthreads();
  }

#pragma unroll
  for (int j = 0; j < 4; ++j)
#pragma unroll
    for (int vv = 0; vv < 8; ++vv) {
      int t = qBase + w * 16 + vv + off;
      int d = j * 16 + col;
      out[(size_t)t * QDIM + head * HD + d] = o[j][vv] / li[vv];
    }
}

// ---------------------------------------------------------------------------
// Router: logits = h @ Wr^T + br ; top-4 softmax -> dense routing [T, 8].
// ---------------------------------------------------------------------------
__global__ __launch_bounds__(256) void router_kernel(
    const float* __restrict__ h, const float* __restrict__ wr,
    const float* __restrict__ br, float* __restrict__ routing, int H) {
  int t = blockIdx.x;
  float p[NEXP] = {};
  for (int i = threadIdx.x; i < H; i += 256) {
    float hv = h[(size_t)t * H + i];
#pragma unroll
    for (int e = 0; e < NEXP; ++e) p[e] += hv * wr[(size_t)e * H + i];
  }
  __shared__ float red[NEXP][256];
#pragma unroll
  for (int e = 0; e < NEXP; ++e) red[e][threadIdx.x] = p[e];
  __syncthreads();
  for (int s = 128; s > 0; s >>= 1) {
    if (threadIdx.x < s)
#pragma unroll
      for (int e = 0; e < NEXP; ++e) red[e][threadIdx.x] += red[e][threadIdx.x + s];
    __syncthreads();
  }
  if (threadIdx.x == 0) {
    float logits[NEXP];
#pragma unroll
    for (int e = 0; e < NEXP; ++e) logits[e] = red[e][0] + br[e];
    int   sel[4];
    float selv[4];
    bool  used[NEXP] = {};
    for (int kk = 0; kk < 4; ++kk) {
      int   bi = 0;
      float bv = -3.4e38f;
      for (int e = 0; e < NEXP; ++e)
        if (!used[e] && logits[e] > bv) { bv = logits[e]; bi = e; }
      used[bi] = true; sel[kk] = bi; selv[kk] = bv;
    }
    float mx = selv[0], den = 0.f, wv[4];
    for (int kk = 0; kk < 4; ++kk) { wv[kk] = __expf(selv[kk] - mx); den += wv[kk]; }
    for (int e = 0; e < NEXP; ++e) routing[(size_t)t * NEXP + e] = 0.f;
    for (int kk = 0; kk < 4; ++kk)
      routing[(size_t)t * NEXP + sel[kk]] = wv[kk] / den;
  }
}

// ---------------------------------------------------------------------------
// SwiGLU (gpt-oss variant, interleaved gate/up columns).
// ---------------------------------------------------------------------------
__global__ void swiglu_kernel(const float* __restrict__ gu,
                              float* __restrict__ act, int T, int I) {
  int idx = blockIdx.x * blockDim.x + threadIdx.x;
  if (idx >= T * I) return;
  int t = idx / I, i = idx % I;
  float g = gu[(size_t)t * 2 * I + 2 * i];
  float u = gu[(size_t)t * 2 * I + 2 * i + 1];
  g = fminf(g, LIMIT);
  u = fminf(fmaxf(u, -LIMIT), LIMIT);
  float sig = 1.0f / (1.0f + __expf(-ALPHA * g));
  act[(size_t)t * I + i] = (u + 1.0f) * (g * sig);
}

// ---------------------------------------------------------------------------
// Host-side orchestration.
// ---------------------------------------------------------------------------
extern "C" void kernel_launch(void* const* d_in, const int* in_sizes, int n_in,
                              void* d_out, int out_size, void* d_ws, size_t ws_size,
                              hipStream_t stream) {
  const float* x        = (const float*)d_in[0];
  const float* ln1_w    = (const float*)d_in[1];
  const float* wq       = (const float*)d_in[2];
  const float* bq       = (const float*)d_in[3];
  const float* wk       = (const float*)d_in[4];
  const float* bk       = (const float*)d_in[5];
  const float* wv       = (const float*)d_in[6];
  const float* bv       = (const float*)d_in[7];
  const float* wo       = (const float*)d_in[8];
  const float* bo       = (const float*)d_in[9];
  const float* sinks    = (const float*)d_in[10];
  const float* ln2_w    = (const float*)d_in[11];
  const float* w_router = (const float*)d_in[12];
  const float* b_router = (const float*)d_in[13];
  const float* w_gate_up = (const float*)d_in[14];
  const float* b_gate_up = (const float*)d_in[15];
  const float* w_down    = (const float*)d_in[16];
  const float* b_down    = (const float*)d_in[17];
  float* out = (float*)d_out;

  // workspace carve-up (floats)
  float* ws = (float*)d_ws;
  size_t off = 0;
  float* h_norm  = ws + off; off += (size_t)S_LEN * HID;
  float* qbuf    = ws + off; off += (size_t)S_LEN * QDIM;
  float* kkbuf   = ws + off; off += (size_t)S_LEN * KVDIM;
  float* vvbuf   = ws + off; off += (size_t)S_LEN * KVDIM;
  float* attn    = ws + off; off += (size_t)S_LEN * QDIM;
  float* x_attn  = ws + off; off += (size_t)S_LEN * HID;
  float* h2      = ws + off; off += (size_t)S_LEN * HID;
  float* routing = ws + off; off += (size_t)S_LEN * NEXP;
  float* gu      = ws + off; off += (size_t)S_LEN * 2 * IDIM;
  float* act     = ws + off; off += (size_t)S_LEN * IDIM;
  (void)ws_size; (void)n_in; (void)in_sizes; (void)out_size;

  // ---- attention sub-block ----
  rmsnorm_kernel<<<S_LEN, 256, 0, stream>>>(x, ln1_w, h_norm, HID);

  gemm_wmma_kernel<<<dim3(QDIM / 64, S_LEN / 128), 256, 0, stream>>>(
      h_norm, wq, bq, nullptr, nullptr, qbuf, S_LEN, QDIM, HID, 0, /*flags=*/1);
  gemm_wmma_kernel<<<dim3(KVDIM / 64, S_LEN / 128), 256, 0, stream>>>(
      h_norm, wk, bk, nullptr, nullptr, kkbuf, S_LEN, KVDIM, HID, 0, 1);
  gemm_wmma_kernel<<<dim3(KVDIM / 64, S_LEN / 128), 256, 0, stream>>>(
      h_norm, wv, bv, nullptr, nullptr, vvbuf, S_LEN, KVDIM, HID, 0, 1);

  {
    int total = S_LEN * (NH + NKV) * 32;
    rope_kernel<<<(total + 255) / 256, 256, 0, stream>>>(qbuf, kkbuf);
  }

  attn_kernel<<<dim3(S_LEN / 64, NH), 128, 0, stream>>>(
      qbuf, kkbuf, vvbuf, sinks, attn);

  // x_attn = x + attn @ wo^T + bo
  gemm_wmma_kernel<<<dim3(HID / 64, S_LEN / 128), 256, 0, stream>>>(
      attn, wo, bo, x, nullptr, x_attn, S_LEN, HID, QDIM, 0, /*flags=*/3);

  // ---- MoE sub-block ----
  rmsnorm_kernel<<<S_LEN, 256, 0, stream>>>(x_attn, ln2_w, h2, HID);
  router_kernel<<<S_LEN, 256, 0, stream>>>(h2, w_router, b_router, routing, HID);

  // out starts as the residual; experts accumulate into it sequentially.
  hipMemcpyAsync(out, x_attn, (size_t)S_LEN * HID * sizeof(float),
                 hipMemcpyDeviceToDevice, stream);

  for (int e = 0; e < NEXP; ++e) {
    const float* wgu = w_gate_up + (size_t)e * 2 * IDIM * HID;
    const float* bgu = b_gate_up + (size_t)e * 2 * IDIM;
    const float* wd  = w_down + (size_t)e * HID * IDIM;
    const float* bd  = b_down + (size_t)e * HID;

    gemm_wmma_kernel<<<dim3(2 * IDIM / 64, S_LEN / 128), 256, 0, stream>>>(
        h2, wgu, bgu, nullptr, nullptr, gu, S_LEN, 2 * IDIM, HID, 0, /*flags=*/1);

    {
      int total = S_LEN * IDIM;
      swiglu_kernel<<<(total + 255) / 256, 256, 0, stream>>>(gu, act, S_LEN, IDIM);
    }

    // out[t,h] += routing[t,e] * (act @ w_down[e]^T + b_down[e])[t,h]
    gemm_wmma_kernel<<<dim3(HID / 64, S_LEN / 128), 256, 0, stream>>>(
        act, wd, bd, nullptr, routing + e, out, S_LEN, HID, IDIM,
        /*scale_stride=*/NEXP, /*flags=*/1 | 4);
  }
}
